// BatchmamutlSwitchExpert_69965017252292
// MI455X (gfx1250) — matrix operand order repaired and checked by reading, not compiled, not run on hardware
//
#include <hip/hip_runtime.h>
#include <hip/hip_bf16.h>

// Problem constants (from reference): E=8, C=4096, D_MODEL=1024, D_FF=4096
#define E_EXPERTS 8
#define C_TOK     4096
#define D_MODEL   1024
#define D_FF      4096

// Tiling: 128x128 block tile, K-slab of 64 staged per barrier round,
// two WMMA k-substeps (16x16x32) per slab.
#define BM  128
#define BN  128
#define BK  64
#define AKP 72   // padded row stride (shorts) for A tile: 144B = 36 dwords -> b128-aligned, conflict-free
#define BKP 72   // padded row stride (shorts) for transposed B tile

typedef __bf16 v16bf __attribute__((ext_vector_type(16)));
typedef float  v8f   __attribute__((ext_vector_type(8)));
typedef float  f32x2 __attribute__((ext_vector_type(2)));
typedef __bf16 bf16x2 __attribute__((ext_vector_type(2)));

union FragAB {
    v16bf    v;
    unsigned u[8];
};

// packed f32x2 -> bf16x2 (RNE) -> 32-bit; lowers to v_cvt_pk_bf16_f32 on gfx1250
static __device__ __forceinline__ unsigned pack_bf16(float a, float b) {
    f32x2  f = {a, b};
    bf16x2 h = __builtin_convertvector(f, bf16x2);
    return __builtin_bit_cast(unsigned, h);
}

// scalar f32 -> bf16 bits, RNE (epilogue only; runs once per block)
static __device__ __forceinline__ unsigned short f2bf_bits(float f) {
    unsigned int u = __builtin_bit_cast(unsigned int, f);
    u += 0x7FFFu + ((u >> 16) & 1u);
    return (unsigned short)(u >> 16);
}

// Generic per-expert tiled GEMM: Out[e] = (relu?)(A[e] * B[e]); A: MxK, B: KxN.
//   A_BF16:   A is bf16 (ushort) else f32 (converted to bf16 while staging to LDS)
//   RELU:     apply relu before store
//   OUT_BF16: store bf16 bits (ushort) else f32
template <bool A_BF16, bool RELU, bool OUT_BF16>
__global__ __launch_bounds__(256) void ffn_gemm_kernel(
    const void* __restrict__ Ain, const float* __restrict__ Bin,
    void* __restrict__ Out, int M, int N, int K) {
    __shared__ __align__(16) unsigned short Alds[BM * AKP];  // [m][k] row-major, padded
    __shared__ __align__(16) unsigned short Btld[BN * BKP];  // [n][k] transposed, padded

    const int e  = blockIdx.z;
    const int mb = blockIdx.y * BM;
    const int nb = blockIdx.x * BN;

    const size_t Abase = (size_t)e * (size_t)M * (size_t)K;
    const size_t Bbase = (size_t)e * (size_t)K * (size_t)N;
    const size_t Obase = (size_t)e * (size_t)M * (size_t)N;

    const int tid  = threadIdx.x;
    const int wave = tid >> 5;
    const int lane = tid & 31;
    const int hl   = lane >> 4;  // ISA half-wave select (lanes 0-15 vs 16-31)
    const int ml   = lane & 15;

    // 8 waves: 4 down M (32 rows each), 2 across N (64 cols each)
    const int wm = (wave & 3) * 32;
    const int wn = (wave >> 2) * 64;

    v8f acc[2][4] = {};

    // staging split (256 threads)
    const int ar  = tid >> 1;        // A row 0..127
    const int ak  = (tid & 1) * 32;  // A k-chunk 0/32
    const int bkq = (tid >> 4) * 4;  // B k-block 0..60
    const int bnq = (tid & 15) * 8;  // B col-block 0..120

    for (int k0 = 0; k0 < K; k0 += BK) {
        // ---- prefetch next K slab (global_prefetch_b8) ----
        if (k0 + BK < K) {
            if (A_BF16)
                __builtin_prefetch((const unsigned short*)Ain + Abase +
                                       (size_t)(mb + ar) * K + (k0 + BK) + ak, 0, 1);
            else
                __builtin_prefetch((const float*)Ain + Abase +
                                       (size_t)(mb + ar) * K + (k0 + BK) + ak, 0, 1);
            __builtin_prefetch(Bin + Bbase + (size_t)(k0 + BK + bkq) * N + nb + bnq, 0, 1);
        }

        // ---- stage A slab: [BM][BK] bf16 (each thread: half a row = 32 elems) ----
        if (A_BF16) {
            const uint4* src = reinterpret_cast<const uint4*>(
                (const unsigned short*)Ain + Abase + (size_t)(mb + ar) * K + k0 + ak);
            uint4* dst = reinterpret_cast<uint4*>(&Alds[ar * AKP + ak]);
#pragma unroll
            for (int i = 0; i < 4; ++i) dst[i] = src[i];
        } else {
            const float* src = (const float*)Ain + Abase + (size_t)(mb + ar) * K + k0 + ak;
            uint4* dst = reinterpret_cast<uint4*>(&Alds[ar * AKP + ak]);
#pragma unroll
            for (int i = 0; i < 4; ++i) {
                float4 v0 = reinterpret_cast<const float4*>(src)[2 * i];
                float4 v1 = reinterpret_cast<const float4*>(src)[2 * i + 1];
                uint4  p;
                p.x = pack_bf16(v0.x, v0.y);
                p.y = pack_bf16(v0.z, v0.w);
                p.z = pack_bf16(v1.x, v1.y);
                p.w = pack_bf16(v1.z, v1.w);
                dst[i] = p;
            }
        }

        // ---- stage B slab transposed: Btld[n][k]; each thread owns a 4k x 8n block,
        //      packs bf16 pairs along k and stores b64s ----
        {
            const float* bs = Bin + Bbase + (size_t)(k0 + bkq) * N + nb + bnq;
            float        rb[4][8];
#pragma unroll
            for (int r = 0; r < 4; ++r) {
                *reinterpret_cast<float4*>(&rb[r][0]) =
                    reinterpret_cast<const float4*>(bs + (size_t)r * N)[0];
                *reinterpret_cast<float4*>(&rb[r][4]) =
                    reinterpret_cast<const float4*>(bs + (size_t)r * N)[1];
            }
#pragma unroll
            for (int c = 0; c < 8; ++c) {
                uint2 p;
                p.x = pack_bf16(rb[0][c], rb[1][c]);
                p.y = pack_bf16(rb[2][c], rb[3][c]);
                *reinterpret_cast<uint2*>(&Btld[(bnq + c) * BKP + bkq]) = p;
            }
        }
        __syncthreads();

        // ---- two k-substeps of 32 per staged slab ----
#pragma unroll
        for (int kk = 0; kk < BK; kk += 32) {
            const int kw = kk >> 1;  // dword offset within a padded row
            FragAB    a[2], b[4];
#pragma unroll
            for (int i = 0; i < 2; ++i) {
                // A 16x32 bf16: lane<16 -> M=ml, K pairs {kw+h*4+j} and {kw+h*4+8+j}
                const unsigned* row =
                    reinterpret_cast<const unsigned*>(&Alds[(wm + 16 * i + ml) * AKP]);
#pragma unroll
                for (int j = 0; j < 4; ++j) {
                    a[i].u[j]     = row[kw + hl * 4 + j];
                    a[i].u[4 + j] = row[kw + hl * 4 + 8 + j];
                }
            }
#pragma unroll
            for (int j = 0; j < 4; ++j) {
                // B 32x16 bf16: lane<16 -> N=ml, K pairs {kw+h*4+t} and {kw+h*4+8+t}
                const unsigned* row =
                    reinterpret_cast<const unsigned*>(&Btld[(wn + 16 * j + ml) * BKP]);
#pragma unroll
                for (int t = 0; t < 4; ++t) {
                    b[j].u[t]     = row[kw + hl * 4 + t];
                    b[j].u[4 + t] = row[kw + hl * 4 + 8 + t];
                }
            }
#pragma unroll
            for (int i = 0; i < 2; ++i)
#pragma unroll
                for (int j = 0; j < 4; ++j)
                    acc[i][j] = __builtin_amdgcn_wmma_f32_16x16x32_bf16(
                        false, a[i].v, false, b[j].v, (short)0, acc[i][j], false, false);
        }
        __syncthreads();
    }

    // ---- epilogue: D 16x16 f32 layout -> row = base + h*8 + r, col = base + ml ----
#pragma unroll
    for (int i = 0; i < 2; ++i) {
#pragma unroll
        for (int j = 0; j < 4; ++j) {
#pragma unroll
            for (int r = 0; r < 8; ++r) {
                int   row = mb + wm + 16 * i + hl * 8 + r;
                int   col = nb + wn + 16 * j + ml;
                float v   = acc[i][j][r];
                if (RELU) v = v > 0.0f ? v : 0.0f;
                if (OUT_BF16)
                    ((unsigned short*)Out)[Obase + (size_t)row * N + col] = f2bf_bits(v);
                else
                    ((float*)Out)[Obase + (size_t)row * N + col] = v;
            }
        }
    }
}

extern "C" void kernel_launch(void* const* d_in, const int* in_sizes, int n_in,
                              void* d_out, int out_size, void* d_ws, size_t ws_size,
                              hipStream_t stream) {
    (void)in_sizes; (void)n_in; (void)out_size; (void)ws_size;

    const float* x   = (const float*)d_in[0];  // [E, C, D_MODEL] f32
    const float* wi  = (const float*)d_in[1];  // [E, D_MODEL, D_FF] f32
    const float* wo  = (const float*)d_in[2];  // [E, D_FF, D_MODEL] f32
    float*       out = (float*)d_out;          // [E*C, D_MODEL] f32

    // Intermediate activation relu(x@wi) as bf16 in workspace: E*C*D_FF*2 = 256 MiB
    unsigned short* act = (unsigned short*)d_ws;

    // Pass 1: act = relu(x @ wi)   [per expert: 4096x1024 @ 1024x4096]
    dim3 g1(D_FF / BN, C_TOK / BM, E_EXPERTS);
    ffn_gemm_kernel<false, true, true><<<g1, 256, 0, stream>>>(
        (const void*)x, wi, (void*)act, C_TOK, D_FF, D_MODEL);

    // Pass 2: out = act @ wo       [per expert: 4096x4096 @ 4096x1024]
    dim3 g2(D_MODEL / BN, C_TOK / BM, E_EXPERTS);
    ffn_gemm_kernel<true, false, false><<<g2, 256, 0, stream>>>(
        (const void*)act, wo, (void*)out, C_TOK, D_MODEL, D_FF);
}
